// LSTMModel_31585189495429
// MI455X (gfx1250) — matrix operand order repaired and checked by reading, not compiled, not run on hardware
//
#include <hip/hip_runtime.h>

// Problem dims (fixed by the reference): B=64, T=512, IN=256, H=512, OUT=256, L=2
#define B_   64
#define T_   512
#define IN_  256
#define H_   512
#define G4H_ (4 * H_)   // 2048
#define OUT_ 256

typedef __bf16 bf16_t;
typedef __attribute__((ext_vector_type(16))) __bf16 v16bf;
typedef __attribute__((ext_vector_type(8)))  __bf16 v8bf;
typedef __attribute__((ext_vector_type(8)))  float  v8f;

// ---------------------------------------------------------------------------
// WMMA fragment loaders (bf16 16x16x32, wave32) per CDNA5 ISA layouts.
// A-matrix 16x32 (MxK), row-major source with leading dim `ld`:
//   lane = half*16 + row ; VGPR v<4 : K = half*8 + 2v{,+1} ; v>=4 : K = 16+half*8+2(v-4){,+1}
//   loaded here in contiguous order to match B: element e : K = half*16 + e
// ---------------------------------------------------------------------------
__device__ __forceinline__ v16bf load_a_frag(const bf16_t* __restrict__ base, int ld, int lane) {
    const int row  = lane & 15;
    const int half = lane >> 4;
    const bf16_t* p = base + (size_t)row * ld + half * 16;
    return *(const v16bf*)(p);          // K = half*16 + [0..16)
}

// B-matrix 32x16 (KxN) with B[k,n] = W[n,k], W row-major [N,K] (k contiguous):
//   lane = half*16 + n ; element e : K = half*16 + e  -> one contiguous 32B load
__device__ __forceinline__ v16bf load_b_frag(const bf16_t* __restrict__ wtile, int ld, int lane) {
    const int n    = lane & 15;
    const int half = lane >> 4;
    return *(const v16bf*)(wtile + (size_t)n * ld + half * 16);
}

// ---------------------------------------------------------------------------
// f32 -> bf16 conversion
// ---------------------------------------------------------------------------
__global__ __launch_bounds__(256) void f32_to_bf16_kernel(const float* __restrict__ in,
                                                          bf16_t* __restrict__ out, int n) {
    int i = blockIdx.x * 256 + threadIdx.x;
    if (i < n) out[i] = (bf16_t)in[i];
}

// ---------------------------------------------------------------------------
// Generic WMMA GEMM: C[M,N] = A_bf16[M,K] @ W_bf16[N,K]^T + bias0[N] (+ bias1[N])
// Block = 128 threads = 4 waves; each wave computes 1 row-tile x 4 col-tiles
// (A fragment reused 4x). grid = (N/256, M/16).
// out_mode 0: C row-major [M,N]   (FC output -> y)
// out_mode 1: m = b*T + t, C[t*(B*4H) + b*4H + n]  (x_proj in [T,B,4H] layout)
// ---------------------------------------------------------------------------
__global__ __launch_bounds__(128) void wmma_gemm_bias(
    const bf16_t* __restrict__ A, const bf16_t* __restrict__ W,
    const float* __restrict__ bias0, const float* __restrict__ bias1,
    float* __restrict__ C, int M, int N, int K, int out_mode)
{
    const int lane = threadIdx.x & 31;
    const int wave = threadIdx.x >> 5;
    const int col  = lane & 15;
    const int half = lane >> 4;
    const int mt   = blockIdx.y;
    const int nt0  = blockIdx.x * 16 + wave * 4;   // first of 4 col-tiles

    v8f acc0 = {}, acc1 = {}, acc2 = {}, acc3 = {};
    const bf16_t* Arow = A + (size_t)(mt * 16) * K;
    const bf16_t* W0 = W + (size_t)((nt0 + 0) * 16) * K;
    const bf16_t* W1 = W + (size_t)((nt0 + 1) * 16) * K;
    const bf16_t* W2 = W + (size_t)((nt0 + 2) * 16) * K;
    const bf16_t* W3 = W + (size_t)((nt0 + 3) * 16) * K;

    for (int k0 = 0; k0 < K; k0 += 32) {
        v16bf a  = load_a_frag(Arow + k0, K, lane);
        v16bf b0 = load_b_frag(W0 + k0, K, lane);
        v16bf b1 = load_b_frag(W1 + k0, K, lane);
        v16bf b2 = load_b_frag(W2 + k0, K, lane);
        v16bf b3 = load_b_frag(W3 + k0, K, lane);
        acc0 = __builtin_amdgcn_wmma_f32_16x16x32_bf16(false, a, false, b0, (short)0, acc0, false, false);
        acc1 = __builtin_amdgcn_wmma_f32_16x16x32_bf16(false, a, false, b1, (short)0, acc1, false, false);
        acc2 = __builtin_amdgcn_wmma_f32_16x16x32_bf16(false, a, false, b2, (short)0, acc2, false, false);
        acc3 = __builtin_amdgcn_wmma_f32_16x16x32_bf16(false, a, false, b3, (short)0, acc3, false, false);
    }

    v8f accs[4] = {acc0, acc1, acc2, acc3};
#pragma unroll
    for (int j = 0; j < 4; ++j) {
        const int n  = (nt0 + j) * 16 + col;
        const float bv = (bias0 ? bias0[n] : 0.f) + (bias1 ? bias1[n] : 0.f);
#pragma unroll
        for (int r = 0; r < 8; ++r) {
            const int m = mt * 16 + half * 8 + r;
            const float v = accs[j][r] + bv;
            if (out_mode == 0) {
                C[(size_t)m * N + n] = v;
            } else {
                const int b = m >> 9;        // m / T_
                const int t = m & (T_ - 1);  // m % T_
                C[(size_t)t * (B_ * G4H_) + (size_t)b * G4H_ + n] = v;
            }
        }
    }
}

// ---------------------------------------------------------------------------
// One LSTM time step (one layer): gates = x_proj[t] + h @ W_hh^T, fused cell.
// grid = H/16 = 32 blocks; block = 512 threads = 16 waves.
// Wave w -> gate g = w>>2, row-tile m = w&3. Each wave: one 16x16 tile, K=512.
// Gates staged in LDS, then cell elementwise. h ping-pong avoids cross-WG races.
// ---------------------------------------------------------------------------
__global__ __launch_bounds__(512) void lstm_step_kernel(
    const bf16_t* __restrict__ Whh,    // [4H, H] bf16 (row-major, k contiguous)
    const float*  __restrict__ xproj,  // [T, B, 4H] f32 (biases already folded in)
    const bf16_t* __restrict__ h_in,   // [B, H] bf16 (unused when is_first)
    float*        __restrict__ c_buf,  // [B, H] f32 state
    bf16_t*       __restrict__ h_next, // [B, H] bf16 (next step's A matrix)
    bf16_t*       __restrict__ seq_out,// [B, T, H] bf16 (feeds next layer / FC)
    float*        __restrict__ hN,     // [B, H] f32 final h (written when is_last)
    float*        __restrict__ cN,     // [B, H] f32 final c (written when is_last)
    int t, int is_first, int is_last)
{
    __shared__ float lds_g[4][B_][16];

    const int lane = threadIdx.x & 31;
    const int wave = threadIdx.x >> 5;
    const int gate = wave >> 2;
    const int mrow = wave & 3;
    const int col  = lane & 15;
    const int half = lane >> 4;
    const int jb   = blockIdx.x;   // H column-tile 0..31

    v8f acc = {};
    if (!is_first) {
        const bf16_t* Arow  = h_in + (size_t)(mrow * 16) * H_;
        const bf16_t* Wtile = Whh + (size_t)((gate * 32 + jb) * 16) * H_;
#pragma unroll 4
        for (int k0 = 0; k0 < H_; k0 += 32) {
            v16bf a = load_a_frag(Arow + k0, H_, lane);
            v16bf b = load_b_frag(Wtile + k0, H_, lane);
            acc = __builtin_amdgcn_wmma_f32_16x16x32_bf16(false, a, false, b, (short)0, acc, false, false);
        }
    }
#pragma unroll
    for (int r = 0; r < 8; ++r)
        lds_g[gate][mrow * 16 + half * 8 + r][col] = acc[r];
    __syncthreads();

    const float* xp = xproj + (size_t)t * (B_ * G4H_);
#pragma unroll
    for (int e = threadIdx.x; e < B_ * 16; e += 512) {
        const int brow = e >> 4;
        const int c    = e & 15;
        const int j    = jb * 16 + c;
        const size_t xb = (size_t)brow * G4H_;
        const float gi = lds_g[0][brow][c] + xp[xb + 0 * H_ + j];
        const float gf = lds_g[1][brow][c] + xp[xb + 1 * H_ + j];
        const float gg = lds_g[2][brow][c] + xp[xb + 2 * H_ + j];
        const float go = lds_g[3][brow][c] + xp[xb + 3 * H_ + j];
        const float i_ = 1.f / (1.f + __expf(-gi));
        const float f_ = 1.f / (1.f + __expf(-gf));
        const float g_ = tanhf(gg);
        const float o_ = 1.f / (1.f + __expf(-go));
        const float cold = is_first ? 0.f : c_buf[brow * H_ + j];
        const float cnew = f_ * cold + i_ * g_;
        const float h    = o_ * tanhf(cnew);
        c_buf[brow * H_ + j] = cnew;
        const bf16_t hb = (bf16_t)h;
        h_next[brow * H_ + j] = hb;
        seq_out[(size_t)brow * (T_ * H_) + (size_t)t * H_ + j] = hb;
        if (is_last) {
            hN[brow * H_ + j] = h;
            cN[brow * H_ + j] = cnew;
        }
    }
}

// ---------------------------------------------------------------------------
extern "C" void kernel_launch(void* const* d_in, const int* in_sizes, int n_in,
                              void* d_out, int out_size, void* d_ws, size_t ws_size,
                              hipStream_t stream)
{
    const float* x    = (const float*)d_in[0];
    const float* Wih0 = (const float*)d_in[1];
    const float* Whh0 = (const float*)d_in[2];
    const float* bih0 = (const float*)d_in[3];
    const float* bhh0 = (const float*)d_in[4];
    const float* Wih1 = (const float*)d_in[5];
    const float* Whh1 = (const float*)d_in[6];
    const float* bih1 = (const float*)d_in[7];
    const float* bhh1 = (const float*)d_in[8];
    const float* Wfc  = (const float*)d_in[9];
    const float* bfc  = (const float*)d_in[10];

    // d_out: y [B,T,OUT] f32, then h_n [2,B,H], then c_n [2,B,H]
    float* y      = (float*)d_out;
    float* hstack = y + (size_t)B_ * T_ * OUT_;
    float* cstack = hstack + 2 * (size_t)B_ * H_;

    // Workspace carve-up (~344 MB total)
    char* ws = (char*)d_ws;
    float*  xproj = (float*)ws;   ws += (size_t)T_ * B_ * G4H_ * sizeof(float);   // 256 MB
    bf16_t* xbf   = (bf16_t*)ws;  ws += (size_t)B_ * T_ * IN_ * 2;                // 16 MB
    bf16_t* out0  = (bf16_t*)ws;  ws += (size_t)B_ * T_ * H_ * 2;                 // 32 MB
    bf16_t* out1  = (bf16_t*)ws;  ws += (size_t)B_ * T_ * H_ * 2;                 // 32 MB
    bf16_t* wih0b = (bf16_t*)ws;  ws += (size_t)G4H_ * IN_ * 2;
    bf16_t* whh0b = (bf16_t*)ws;  ws += (size_t)G4H_ * H_ * 2;
    bf16_t* wih1b = (bf16_t*)ws;  ws += (size_t)G4H_ * H_ * 2;
    bf16_t* whh1b = (bf16_t*)ws;  ws += (size_t)G4H_ * H_ * 2;
    bf16_t* wfcb  = (bf16_t*)ws;  ws += (size_t)OUT_ * H_ * 2;
    bf16_t* hA    = (bf16_t*)ws;  ws += (size_t)B_ * H_ * 2;
    bf16_t* hB    = (bf16_t*)ws;  ws += (size_t)B_ * H_ * 2;
    float*  cbuf  = (float*)ws;   ws += (size_t)B_ * H_ * sizeof(float);

    auto cvt = [&](const float* src, bf16_t* dst, int n) {
        f32_to_bf16_kernel<<<dim3((n + 255) / 256), dim3(256), 0, stream>>>(src, dst, n);
    };
    cvt(x,    xbf,   B_ * T_ * IN_);
    cvt(Wih0, wih0b, G4H_ * IN_);
    cvt(Whh0, whh0b, G4H_ * H_);
    cvt(Wih1, wih1b, G4H_ * H_);
    cvt(Whh1, whh1b, G4H_ * H_);
    cvt(Wfc,  wfcb,  OUT_ * H_);

    const int M = B_ * T_;                  // 32768
    bf16_t* hbuf[2] = {hA, hB};

    // ---- Layer 0 ----
    wmma_gemm_bias<<<dim3(G4H_ / 256, M / 16), dim3(128), 0, stream>>>(
        xbf, wih0b, bih0, bhh0, xproj, M, G4H_, IN_, /*out_mode=*/1);
    for (int t = 0; t < T_; ++t) {
        lstm_step_kernel<<<dim3(H_ / 16), dim3(512), 0, stream>>>(
            whh0b, xproj, hbuf[t & 1], cbuf, hbuf[(t & 1) ^ 1], out0,
            hstack + 0 * (size_t)B_ * H_, cstack + 0 * (size_t)B_ * H_,
            t, (t == 0) ? 1 : 0, (t == T_ - 1) ? 1 : 0);
    }

    // ---- Layer 1 ----
    wmma_gemm_bias<<<dim3(G4H_ / 256, M / 16), dim3(128), 0, stream>>>(
        out0, wih1b, bih1, bhh1, xproj, M, G4H_, H_, /*out_mode=*/1);
    for (int t = 0; t < T_; ++t) {
        lstm_step_kernel<<<dim3(H_ / 16), dim3(512), 0, stream>>>(
            whh1b, xproj, hbuf[t & 1], cbuf, hbuf[(t & 1) ^ 1], out1,
            hstack + 1 * (size_t)B_ * H_, cstack + 1 * (size_t)B_ * H_,
            t, (t == 0) ? 1 : 0, (t == T_ - 1) ? 1 : 0);
    }

    // ---- FC: y = out1 @ W_fc^T + b_fc ----
    wmma_gemm_bias<<<dim3(OUT_ / 256, M / 16), dim3(128), 0, stream>>>(
        out1, wfcb, bfc, nullptr, y, M, OUT_, H_, /*out_mode=*/0);
}
